// CostVolume_57853209477696
// MI455X (gfx1250) — matrix operand order repaired and checked by reading, not compiled, not run on hardware
//
#include <hip/hip_runtime.h>
#include <hip/hip_bf16.h>

// ---------------------------------------------------------------------------
// CostVolume pipeline for MI455X (gfx1250, wave32, WMMA bf16 16x16x32)
//   downsample (1x1 conv GEMM)  -> feat bf16 [img][h][w][64]   (BN1+ReLU fused)
//   conv3d #1 (implicit GEMM, cost volume gathered on the fly)
//                               -> mid  bf16 [d][h][w][64]     (BN2+ReLU fused)
//   conv3d #2 (implicit GEMM)   -> out  f32  [(c*48+d)*H*W...] (BN3+ReLU fused)
// Conv kernels: 64oc x 64w tile, 2 WMMA per wave per kstep (A reuse).
// Interior blocks: triple-buffered async global->LDS pipeline
//   (global_load_async_to_lds_b128 issued 2 ksteps ahead, s_wait_asynccnt 1).
// Boundary blocks: masked manual loads, double-buffered, 1 barrier/kstep.
// ---------------------------------------------------------------------------

typedef __bf16 v16bf __attribute__((ext_vector_type(16)));
typedef float  v8f   __attribute__((ext_vector_type(8)));

#define CIN   1024
#define CF    64
#define DD    48
#define HH    96
#define WW    320
#define HW    (HH * WW)
#define TILE  (64 * 32)   // B-tile elements per buffer

__device__ __forceinline__ unsigned short f2bf(float f) {
  __bf16 h = (__bf16)f;
  return __builtin_bit_cast(unsigned short, h);
}

// Async copy 16B global -> LDS (per lane), tracked by ASYNCcnt.
__device__ __forceinline__ void async_ld16(unsigned lds_addr,
                                           unsigned long long gaddr) {
  asm volatile("global_load_async_to_lds_b128 %0, %1, off"
               :: "v"(lds_addr), "v"(gaddr) : "memory");
}
__device__ __forceinline__ void wait_async(int pending) {
  if (pending)
    asm volatile("s_wait_asynccnt 0x1" ::: "memory");
  else
    asm volatile("s_wait_asynccnt 0x0" ::: "memory");
}

// ---------------------------------------------------------------------------
// BN fold: s = g * rsqrt(v+eps); t = (conv_bias - mean) * s + beta
// sbt layout: [which][ s[64] | t[64] ]  (which: 0=bn1, 1=bn2, 2=bn3)
// ---------------------------------------------------------------------------
__global__ void k_prep_bn(const float* __restrict__ b_ds,
                          const float* __restrict__ g1, const float* __restrict__ be1,
                          const float* __restrict__ m1, const float* __restrict__ v1,
                          const float* __restrict__ b1c,
                          const float* __restrict__ g2, const float* __restrict__ be2,
                          const float* __restrict__ m2, const float* __restrict__ v2,
                          const float* __restrict__ b2c,
                          const float* __restrict__ g3, const float* __restrict__ be3,
                          const float* __restrict__ m3, const float* __restrict__ v3,
                          float* __restrict__ sbt) {
  int t = threadIdx.x;
  if (t >= 192) return;
  int ch = t & 63, which = t >> 6;
  const float *g, *be, *mm, *vv, *bias;
  if (which == 0)      { g = g1; be = be1; mm = m1; vv = v1; bias = b_ds; }
  else if (which == 1) { g = g2; be = be2; mm = m2; vv = v2; bias = b1c; }
  else                 { g = g3; be = be3; mm = m3; vv = v3; bias = b2c; }
  float s  = g[ch] * rsqrtf(vv[ch] + 1e-3f);
  float tt = (bias[ch] - mm[ch]) * s + be[ch];
  sbt[which * 128 + ch]      = s;
  sbt[which * 128 + 64 + ch] = tt;
}

// ---------------------------------------------------------------------------
// Weight swizzle into per-lane WMMA A fragments (16-bit A 16x32), contiguous k order (same as B):
//   lanes 0..15 : M=lane,    slots 0..15 -> K=0..15
//   lanes 16..31: M=lane-16, slots 0..15 -> K=16..31
// aswz[kstep][mtile][lane][slot], 32B contiguous per lane.
// ---------------------------------------------------------------------------
__global__ void k_prep_swz(const float* __restrict__ w_ds,
                           const float* __restrict__ w1,
                           const float* __restrict__ w2,
                           unsigned short* __restrict__ a0,
                           unsigned short* __restrict__ a1,
                           unsigned short* __restrict__ a2) {
  const int N0 = 32  * 4 * 32 * 16;   // 65536   (K=1024)
  const int N1 = 108 * 4 * 32 * 16;   // 221184  (K=27*128)
  const int N2 = 54  * 4 * 32 * 16;   // 110592  (K=27*64)
  int i = blockIdx.x * 256 + threadIdx.x;
  if (i >= N0 + N1 + N2) return;
  int region, j;
  if (i < N0)           { region = 0; j = i; }
  else if (i < N0 + N1) { region = 1; j = i - N0; }
  else                  { region = 2; j = i - N0 - N1; }

  int slot  = j & 15;
  int lane  = (j >> 4) & 31;
  int mtile = (j >> 9) & 3;
  int kstep = j >> 11;
  int klocal = slot + ((lane & 16) ? 16 : 0);
  int m = mtile * 16 + (lane & 15);
  int k = kstep * 32 + klocal;

  if (region == 0) {
    a0[j] = f2bf(w_ds[m * CIN + k]);                       // w_ds[oc][cin]
  } else if (region == 1) {
    int tap = k >> 7, ic = k & 127;                        // K = tap*128 + ic
    int kd = tap / 9, kh = (tap / 3) % 3, kw = tap % 3;
    a1[j] = f2bf(w1[(((size_t)(m * 128 + ic) * 3 + kd) * 3 + kh) * 3 + kw]);
  } else {
    int tap = k >> 6, ic = k & 63;                         // K = tap*64 + ic
    int kd = tap / 9, kh = (tap / 3) % 3, kw = tap % 3;
    a2[j] = f2bf(w2[(((size_t)(m * 64 + ic) * 3 + kd) * 3 + kh) * 3 + kw]);
  }
}

// ---------------------------------------------------------------------------
// B-fragment LDS read (16-bit B 32x16): lane n(0..15) holds column N=n, K=0..15;
// lane n+16 holds column N=n, K=16..31. Tile stored [n][k(32)] -> 32B/lane.
// ---------------------------------------------------------------------------
__device__ __forceinline__ v16bf load_bfrag(const unsigned short* Bt, int ntile, int lane) {
  return *(const v16bf*)(&Bt[(ntile * 16 + (lane & 15)) * 32 + (lane >> 4) * 16]);
}

// ---------------------------------------------------------------------------
// Downsample GEMM: [64 x 1024] x [1024 x 30720] per image, BN1+ReLU,
// store bf16 channels-last feat[img][h*W+w][64].
// Grid: (960 position-tiles, 2 images), 256 threads.
// ---------------------------------------------------------------------------
__global__ void k_downsample(const float* __restrict__ lf,
                             const float* __restrict__ rf,
                             const unsigned short* __restrict__ a0,
                             const float* __restrict__ sbt,
                             unsigned short* __restrict__ feat) {
  const int img = blockIdx.y;
  const int p0  = blockIdx.x * 32;
  const float* src = img ? rf : lf;
  unsigned short* dst = feat + (size_t)img * HW * CF;

  __shared__ __align__(16) unsigned short Bt[32 * 32];

  const int t = threadIdx.x;
  const int lane = t & 31, wave = t >> 5;
  const int mtile = wave & 3, ntile = wave >> 2;
  const int cl = t >> 3;          // klocal 0..31
  const int nq = (t & 7) * 4;     // n offset

  v8f acc = {0.f, 0.f, 0.f, 0.f, 0.f, 0.f, 0.f, 0.f};

  for (int ks = 0; ks < 32; ++ks) {
    float4 v = *(const float4*)(src + (size_t)(ks * 32 + cl) * HW + p0 + nq);
    __syncthreads();
    Bt[(nq + 0) * 32 + cl] = f2bf(v.x);
    Bt[(nq + 1) * 32 + cl] = f2bf(v.y);
    Bt[(nq + 2) * 32 + cl] = f2bf(v.z);
    Bt[(nq + 3) * 32 + cl] = f2bf(v.w);
    __syncthreads();

    v16bf a = *(const v16bf*)(a0 + ((size_t)(ks * 4 + mtile) * 32 + lane) * 16);
    if (ks + 1 < 32)
      __builtin_prefetch(a0 + ((size_t)((ks + 1) * 4 + mtile) * 32 + lane) * 16, 0, 3);
    v16bf b = load_bfrag(Bt, ntile, lane);
    acc = __builtin_amdgcn_wmma_f32_16x16x32_bf16(false, a, false, b,
                                                  (short)0, acc, false, false);
  }

  // Epilogue: BN1+ReLU, 8 contiguous oc per lane -> one 16B store.
  const int n   = p0 + ntile * 16 + (lane & 15);
  const int ocb = mtile * 16 + ((lane & 16) ? 8 : 0);
  const float* S = sbt;  // which=0
  __align__(16) unsigned short o[8];
#pragma unroll
  for (int r = 0; r < 8; ++r) {
    int oc = ocb + r;
    float v = acc[r] * S[oc] + S[64 + oc];
    o[r] = f2bf(v > 0.f ? v : 0.f);
  }
  *(uint4*)(dst + (size_t)n * CF + ocb) = *(const uint4*)o;
}

// ---------------------------------------------------------------------------
// Per-kstep gather address for conv1 (cost-volume fused) and conv2.
// ---------------------------------------------------------------------------
struct GAddr { const unsigned short* sp; bool ok; };

__device__ __forceinline__ GAddr c1_addr(const unsigned short* feat,
                                         int d, int h, int w0, int nl, int icl,
                                         int ks) {
  int tap  = ks >> 2;
  int kd   = tap / 9, kh = (tap / 3) % 3, kw = tap % 3;
  int sub  = ks & 3;
  int half = sub >> 1;          // 0 = left, 1 = right(shifted)
  int icb  = (sub & 1) * 32;
  int dp = d + kd - 1;
  int hp = h + kh - 1;
  int wp = w0 + nl + kw - 1;
  int wsrc = half ? (wp - dp) : wp;
  GAddr r;
  r.ok = ((unsigned)dp < (unsigned)DD) && ((unsigned)hp < (unsigned)HH) &&
         ((unsigned)wp < (unsigned)WW) && (wp >= dp);
  r.sp = feat + ((size_t)half * HW + (size_t)hp * WW + wsrc) * CF + icb + icl;
  return r;
}

__device__ __forceinline__ GAddr c2_addr(const unsigned short* mid,
                                         int d, int h, int w0, int nl, int icl,
                                         int ks) {
  int tap = ks >> 1;
  int kd  = tap / 9, kh = (tap / 3) % 3, kw = tap % 3;
  int icb = (ks & 1) * 32;
  int dp = d + kd - 1;
  int hp = h + kh - 1;
  int wp = w0 + nl + kw - 1;
  GAddr r;
  r.ok = ((unsigned)dp < (unsigned)DD) && ((unsigned)hp < (unsigned)HH) &&
         ((unsigned)wp < (unsigned)WW);
  r.sp = mid + (((size_t)dp * HH + hp) * WW + wp) * CF + icb + icl;
  return r;
}

// ---------------------------------------------------------------------------
// conv3d #1 as implicit GEMM with fused cost-volume gather.
// M=64 oc, K=27 taps * 128 ic (108 ksteps of 32), N=64 w at fixed (d,h).
// Grid: (5 wtiles, 96 h, 48 d), 256 threads.
// ---------------------------------------------------------------------------
__global__ void k_conv1(const unsigned short* __restrict__ feat,
                        const unsigned short* __restrict__ a1,
                        const float* __restrict__ sbt,
                        unsigned short* __restrict__ mid) {
  const int w0 = blockIdx.x * 64;
  const int h  = blockIdx.y;
  const int d  = blockIdx.z;
  const int KS = 108;

  __shared__ __align__(16) unsigned short Bt[3 * TILE];

  const int t = threadIdx.x;
  const int lane = t & 31, wave = t >> 5;
  const int mtile = wave & 3, nh = wave >> 2;
  const int nl  = t >> 2;         // n 0..63
  const int icl = (t & 3) * 8;    // 8 contiguous channels (16B)
  const int lslot = nl * 32 + icl;

  // Block-uniform: every (nl, tap) load in-bounds and disparity-mask valid.
  const bool interior = (d >= 1) && (d <= DD - 2) && (h >= 1) && (h <= HH - 2) &&
                        (w0 >= 64) && (w0 + 65 <= WW - 1);

  v8f acc0 = {0.f, 0.f, 0.f, 0.f, 0.f, 0.f, 0.f, 0.f};
  v8f acc1 = {0.f, 0.f, 0.f, 0.f, 0.f, 0.f, 0.f, 0.f};

  if (interior) {
    // Triple-buffered async pipeline: issue kstep+2, wait for kstep, 1 barrier.
    async_ld16((unsigned)(size_t)&Bt[0 * TILE + lslot],
               (unsigned long long)(size_t)c1_addr(feat, d, h, w0, nl, icl, 0).sp);
    async_ld16((unsigned)(size_t)&Bt[1 * TILE + lslot],
               (unsigned long long)(size_t)c1_addr(feat, d, h, w0, nl, icl, 1).sp);
    for (int ks = 0; ks < KS; ++ks) {
      wait_async(ks + 1 < KS);
      __syncthreads();
      if (ks + 2 < KS)
        async_ld16((unsigned)(size_t)&Bt[((ks + 2) % 3) * TILE + lslot],
                   (unsigned long long)(size_t)c1_addr(feat, d, h, w0, nl, icl, ks + 2).sp);
      const unsigned short* buf = &Bt[(ks % 3) * TILE];
      v16bf a = *(const v16bf*)(a1 + ((size_t)(ks * 4 + mtile) * 32 + lane) * 16);
      if (ks + 1 < KS)
        __builtin_prefetch(a1 + ((size_t)((ks + 1) * 4 + mtile) * 32 + lane) * 16, 0, 3);
      v16bf b0 = load_bfrag(buf, nh * 2 + 0, lane);
      v16bf b1 = load_bfrag(buf, nh * 2 + 1, lane);
      acc0 = __builtin_amdgcn_wmma_f32_16x16x32_bf16(false, a, false, b0,
                                                     (short)0, acc0, false, false);
      acc1 = __builtin_amdgcn_wmma_f32_16x16x32_bf16(false, a, false, b1,
                                                     (short)0, acc1, false, false);
    }
  } else {
    // Masked manual path, double-buffered, one barrier per kstep.
    for (int ks = 0; ks < KS; ++ks) {
      GAddr ga = c1_addr(feat, d, h, w0, nl, icl, ks);
      uint4 val = {0u, 0u, 0u, 0u};
      if (ga.ok) val = *(const uint4*)ga.sp;
      unsigned short* buf = &Bt[(ks & 1) * TILE];
      *(uint4*)&buf[lslot] = val;
      __syncthreads();
      v16bf a = *(const v16bf*)(a1 + ((size_t)(ks * 4 + mtile) * 32 + lane) * 16);
      v16bf b0 = load_bfrag(buf, nh * 2 + 0, lane);
      v16bf b1 = load_bfrag(buf, nh * 2 + 1, lane);
      acc0 = __builtin_amdgcn_wmma_f32_16x16x32_bf16(false, a, false, b0,
                                                     (short)0, acc0, false, false);
      acc1 = __builtin_amdgcn_wmma_f32_16x16x32_bf16(false, a, false, b1,
                                                     (short)0, acc1, false, false);
    }
  }

  // Epilogue: BN2+ReLU, bf16 channels-last mid[d][h][w][64].
  const int ocb = mtile * 16 + ((lane & 16) ? 8 : 0);
  const float* S = sbt + 128;
#pragma unroll
  for (int j = 0; j < 2; ++j) {
    const v8f& acc = j ? acc1 : acc0;
    int n = w0 + nh * 32 + j * 16 + (lane & 15);
    __align__(16) unsigned short o[8];
#pragma unroll
    for (int r = 0; r < 8; ++r) {
      int oc = ocb + r;
      float v = acc[r] * S[oc] + S[64 + oc];
      o[r] = f2bf(v > 0.f ? v : 0.f);
    }
    *(uint4*)(mid + (((size_t)d * HH + h) * WW + n) * CF + ocb) = *(const uint4*)o;
  }
}

// ---------------------------------------------------------------------------
// conv3d #2 as implicit GEMM. M=64, K=27*64 (54 ksteps), N=64 w.
// Output f32 with BN3+ReLU into d_out[(oc*48+d)*H*W + h*W + w].
// ---------------------------------------------------------------------------
__global__ void k_conv2(const unsigned short* __restrict__ mid,
                        const unsigned short* __restrict__ a2,
                        const float* __restrict__ sbt,
                        float* __restrict__ out) {
  const int w0 = blockIdx.x * 64;
  const int h  = blockIdx.y;
  const int d  = blockIdx.z;
  const int KS = 54;

  __shared__ __align__(16) unsigned short Bt[3 * TILE];

  const int t = threadIdx.x;
  const int lane = t & 31, wave = t >> 5;
  const int mtile = wave & 3, nh = wave >> 2;
  const int nl  = t >> 2;
  const int icl = (t & 3) * 8;
  const int lslot = nl * 32 + icl;

  const bool interior = (d >= 1) && (d <= DD - 2) && (h >= 1) && (h <= HH - 2) &&
                        (w0 >= 64) && (w0 + 65 <= WW - 1);

  v8f acc0 = {0.f, 0.f, 0.f, 0.f, 0.f, 0.f, 0.f, 0.f};
  v8f acc1 = {0.f, 0.f, 0.f, 0.f, 0.f, 0.f, 0.f, 0.f};

  if (interior) {
    async_ld16((unsigned)(size_t)&Bt[0 * TILE + lslot],
               (unsigned long long)(size_t)c2_addr(mid, d, h, w0, nl, icl, 0).sp);
    async_ld16((unsigned)(size_t)&Bt[1 * TILE + lslot],
               (unsigned long long)(size_t)c2_addr(mid, d, h, w0, nl, icl, 1).sp);
    for (int ks = 0; ks < KS; ++ks) {
      wait_async(ks + 1 < KS);
      __syncthreads();
      if (ks + 2 < KS)
        async_ld16((unsigned)(size_t)&Bt[((ks + 2) % 3) * TILE + lslot],
                   (unsigned long long)(size_t)c2_addr(mid, d, h, w0, nl, icl, ks + 2).sp);
      const unsigned short* buf = &Bt[(ks % 3) * TILE];
      v16bf a = *(const v16bf*)(a2 + ((size_t)(ks * 4 + mtile) * 32 + lane) * 16);
      if (ks + 1 < KS)
        __builtin_prefetch(a2 + ((size_t)((ks + 1) * 4 + mtile) * 32 + lane) * 16, 0, 3);
      v16bf b0 = load_bfrag(buf, nh * 2 + 0, lane);
      v16bf b1 = load_bfrag(buf, nh * 2 + 1, lane);
      acc0 = __builtin_amdgcn_wmma_f32_16x16x32_bf16(false, a, false, b0,
                                                     (short)0, acc0, false, false);
      acc1 = __builtin_amdgcn_wmma_f32_16x16x32_bf16(false, a, false, b1,
                                                     (short)0, acc1, false, false);
    }
  } else {
    for (int ks = 0; ks < KS; ++ks) {
      GAddr ga = c2_addr(mid, d, h, w0, nl, icl, ks);
      uint4 val = {0u, 0u, 0u, 0u};
      if (ga.ok) val = *(const uint4*)ga.sp;
      unsigned short* buf = &Bt[(ks & 1) * TILE];
      *(uint4*)&buf[lslot] = val;
      __syncthreads();
      v16bf a = *(const v16bf*)(a2 + ((size_t)(ks * 4 + mtile) * 32 + lane) * 16);
      v16bf b0 = load_bfrag(buf, nh * 2 + 0, lane);
      v16bf b1 = load_bfrag(buf, nh * 2 + 1, lane);
      acc0 = __builtin_amdgcn_wmma_f32_16x16x32_bf16(false, a, false, b0,
                                                     (short)0, acc0, false, false);
      acc1 = __builtin_amdgcn_wmma_f32_16x16x32_bf16(false, a, false, b1,
                                                     (short)0, acc1, false, false);
    }
  }

  // Epilogue: BN3+ReLU, f32 output, out[(oc*48+d)*HW + h*W + w].
  const int ocb = mtile * 16 + ((lane & 16) ? 8 : 0);
  const float* S = sbt + 256;
#pragma unroll
  for (int j = 0; j < 2; ++j) {
    const v8f& acc = j ? acc1 : acc0;
    int n = w0 + nh * 32 + j * 16 + (lane & 15);
#pragma unroll
    for (int r = 0; r < 8; ++r) {
      int oc = ocb + r;
      float v = acc[r] * S[oc] + S[64 + oc];
      v = v > 0.f ? v : 0.f;
      out[((size_t)(oc * DD + d) * HH + h) * WW + n] = v;
    }
  }
}

// ---------------------------------------------------------------------------
// Host launcher. d_ws layout (bytes):
//   [0,1536)            sbt (3 * 128 f32)
//   [2048, +131072)     a0  (swizzled w_ds)
//   [133120, +442368)   a1  (swizzled w3d_1)
//   [575488, +221184)   a2  (swizzled w3d_2)
//   [796672, +7864320)  feat bf16 [2][96*320][64]
//   [8660992, +188743680) mid bf16 [48][96*320][64]
// ---------------------------------------------------------------------------
extern "C" void kernel_launch(void* const* d_in, const int* in_sizes, int n_in,
                              void* d_out, int out_size, void* d_ws, size_t ws_size,
                              hipStream_t stream) {
  const float* left  = (const float*)d_in[0];
  const float* right = (const float*)d_in[1];
  const float* w_ds  = (const float*)d_in[2];
  const float* b_ds  = (const float*)d_in[3];
  const float* bn1_g = (const float*)d_in[4];
  const float* bn1_b = (const float*)d_in[5];
  const float* bn1_m = (const float*)d_in[6];
  const float* bn1_v = (const float*)d_in[7];
  const float* w3d_1 = (const float*)d_in[8];
  const float* b3d_1 = (const float*)d_in[9];
  const float* bn2_g = (const float*)d_in[10];
  const float* bn2_b = (const float*)d_in[11];
  const float* bn2_m = (const float*)d_in[12];
  const float* bn2_v = (const float*)d_in[13];
  const float* w3d_2 = (const float*)d_in[14];
  const float* b3d_2 = (const float*)d_in[15];
  const float* bn3_g = (const float*)d_in[16];
  const float* bn3_b = (const float*)d_in[17];
  const float* bn3_m = (const float*)d_in[18];
  const float* bn3_v = (const float*)d_in[19];

  char* ws = (char*)d_ws;
  float*          sbt  = (float*)ws;
  unsigned short* a0   = (unsigned short*)(ws + 2048);
  unsigned short* a1   = (unsigned short*)(ws + 133120);
  unsigned short* a2   = (unsigned short*)(ws + 575488);
  unsigned short* feat = (unsigned short*)(ws + 796672);
  unsigned short* mid  = (unsigned short*)(ws + 8660992);

  k_prep_bn<<<1, 256, 0, stream>>>(b_ds, bn1_g, bn1_b, bn1_m, bn1_v,
                                   b3d_1, bn2_g, bn2_b, bn2_m, bn2_v,
                                   b3d_2, bn3_g, bn3_b, bn3_m, bn3_v, sbt);

  const int total_swz = 65536 + 221184 + 110592;
  k_prep_swz<<<(total_swz + 255) / 256, 256, 0, stream>>>(w_ds, w3d_1, w3d_2,
                                                          a0, a1, a2);

  k_downsample<<<dim3(HW / 32, 2), 256, 0, stream>>>(left, right, a0, sbt, feat);

  k_conv1<<<dim3(WW / 64, HH, DD), 256, 0, stream>>>(feat, a1, sbt, mid);

  k_conv2<<<dim3(WW / 64, HH, DD), 256, 0, stream>>>(mid, a2, sbt,
                                                     (float*)d_out);
}
